// MultiHeadAttention_ttnn_24421184045269
// MI455X (gfx1250) — matrix operand order repaired and checked by reading, not compiled, not run on hardware
//
#include <hip/hip_runtime.h>

typedef __attribute__((ext_vector_type(16))) __bf16        v16bf;
typedef __attribute__((ext_vector_type(8)))  float         v8f;
typedef __attribute__((ext_vector_type(4)))  unsigned int  u32x4;
typedef __attribute__((ext_vector_type(4)))  float         f32x4;
typedef __attribute__((ext_vector_type(8)))  int           i32x8;
typedef __attribute__((ext_vector_type(4)))  int           i32x4;

#define D_MODEL 1024
#define SEQ     2048
#define BATCH   2
#define NHEAD   16
#define HDIM    64

__device__ __forceinline__ unsigned short f2bf(float f) {
    unsigned int u = __float_as_uint(f);
    u += 0x7fffu + ((u >> 16) & 1u);          // round-to-nearest-even
    return (unsigned short)(u >> 16);
}

__device__ __forceinline__ v8f vzero8() {
    v8f z;
    for (int i = 0; i < 8; ++i) z[i] = 0.0f;
    return z;
}

union Frag {
    u32x4 u[2];
    v16bf v;
};

// A-matrix 16x32 bf16 fragment from a row-major ushort tile.
// Lanes 0-15: K chunks [0..7] and [16..23]; lanes 16-31: [8..15] and [24..31].
__device__ __forceinline__ v16bf load_a_frag(const unsigned short* tile, int stride, int lane) {
    Frag f;
    const int m    = lane & 15;
    const int half = lane >> 4;
    const unsigned short* p = tile + m * stride + half * 8;
    f.u[0] = *(const u32x4*)(p);
    f.u[1] = *(const u32x4*)(p + 16);
    return f.v;
}

// B-matrix 32x16 bf16 fragment where B[k][n] = M[n][k] with M row-major:
// Lanes 0-15: K chunks [0..7] and [16..23]; lanes 16-31: [8..15] and [24..31].
__device__ __forceinline__ v16bf load_b_frag(const unsigned short* m_tile, int stride, int lane) {
    Frag f;
    const int n  = lane & 15;
    const int kc = (lane >> 4) * 8;
    const unsigned short* p = m_tile + n * stride + kc;
    f.u[0] = *(const u32x4*)(p);
    f.u[1] = *(const u32x4*)(p + 16);
    return f.v;
}

// ---------------------------------------------------------------------------
// Tensor Data Mover: 2D tile load (bf16, data_size=2B) global -> LDS.
// D# layout per CDNA5 ISA. Groups 2/3 zero (2D tensor).
// 6-arg builtin variant (clang-23 / therock-10.0 headers).
// ---------------------------------------------------------------------------
__device__ __forceinline__ void tdm_load_2d(unsigned lds_off, const unsigned short* gptr,
                                            int tile_cols, int tile_rows,
                                            int tensor_cols, int tensor_rows,
                                            int row_stride_elems) {
  unsigned long long ga = (unsigned long long)(size_t)gptr;
  unsigned ga_lo = (unsigned)ga;
  unsigned ga_hi = (unsigned)(ga >> 32);

  u32x4 g0;
  g0[0] = 1u;                                                   // count=1, user D#
  g0[1] = (unsigned)__builtin_amdgcn_readfirstlane(lds_off);    // lds_addr [63:32]
  g0[2] = (unsigned)__builtin_amdgcn_readfirstlane(ga_lo);      // global_addr[31:0]
  g0[3] = ((unsigned)__builtin_amdgcn_readfirstlane(ga_hi) & 0x1ffffffu)
          | (2u << 30);                                         // addr[56:32], type=2

  i32x8 g1;
  g1[0] = (int)(1u << 16);                                      // data_size=1 (2 bytes)
  g1[1] = (int)(((unsigned)tensor_cols & 0xffffu) << 16);       // tensor_dim0[15:0]
  g1[2] = (int)((((unsigned)tensor_cols >> 16) & 0xffffu)
          | (((unsigned)tensor_rows & 0xffffu) << 16));         // dim0 hi | dim1 lo
  g1[3] = (int)((((unsigned)tensor_rows >> 16) & 0xffffu)
          | (((unsigned)tile_cols & 0xffffu) << 16));           // dim1 hi | tile_dim0
  g1[4] = (int)((unsigned)tile_rows & 0xffffu);                 // tile_dim1, tile_dim2=0
  g1[5] = (int)(unsigned)row_stride_elems;                      // tensor_dim0_stride lo
  g1[6] = 0;
  g1[7] = 0;
  for (int i = 0; i < 8; ++i) g1[i] = __builtin_amdgcn_readfirstlane(g1[i]);

  i32x4 z4;
  z4[0] = 0; z4[1] = 0; z4[2] = 0; z4[3] = 0;
  i32x8 z8;
  for (int i = 0; i < 8; ++i) z8[i] = 0;
  __builtin_amdgcn_tensor_load_to_lds(g0, g1, z4, z4, z8, 0);
}

__device__ __forceinline__ unsigned lds_offset_of(const void* p) {
  // generic address of a __shared__ object: addr[31:0] == LDS byte offset
  return (unsigned)(size_t)p;
}

// ---------------------------------------------------------------------------
// Kernel 0: fp32 -> bf16 bulk conversion (vectorized, 8 elems/thread/iter)
// ---------------------------------------------------------------------------
__global__ __launch_bounds__(256) void cvt_bf16_kernel(const float* __restrict__ src,
                                                       unsigned short* __restrict__ dst,
                                                       int n) {
  int i = (blockIdx.x * 256 + threadIdx.x) * 8;
  const int stride = gridDim.x * 256 * 8;
  for (; i < n; i += stride) {
    f32x4 a = *(const f32x4*)(src + i);
    f32x4 b = *(const f32x4*)(src + i + 4);
    u32x4 o;
    o[0] = (unsigned)f2bf(a[0]) | ((unsigned)f2bf(a[1]) << 16);
    o[1] = (unsigned)f2bf(a[2]) | ((unsigned)f2bf(a[3]) << 16);
    o[2] = (unsigned)f2bf(b[0]) | ((unsigned)f2bf(b[1]) << 16);
    o[3] = (unsigned)f2bf(b[2]) | ((unsigned)f2bf(b[3]) << 16);
    *(u32x4*)(dst + i) = o;
  }
}

// ---------------------------------------------------------------------------
// Kernel 1: Q/K/V = xb @ W^T (all bf16), TDM-staged LDS tiles, double buffered.
// WG = 256 threads = 8 waves; tile 128 rows x 64 cols; K-step 64.
// ---------------------------------------------------------------------------
__global__ __launch_bounds__(256) void qkv_proj_kernel(
    const unsigned short* __restrict__ xb,
    const unsigned short* __restrict__ wqb,
    const unsigned short* __restrict__ wkb,
    const unsigned short* __restrict__ wvb,
    unsigned short* __restrict__ qkv_out) {
  __shared__ unsigned short bufA[2][128 * 64];   // 32 KB
  __shared__ unsigned short bufW[2][64 * 64];    // 16 KB
  const int tid  = threadIdx.x;
  const int lane = tid & 31;
  const int wave = tid >> 5;
  const int wm   = wave & 3;
  const int wn   = wave >> 2;
  const int rowBase = blockIdx.x * 128;
  const int colBase = blockIdx.y * 64;
  const unsigned short* W =
      (blockIdx.z == 0) ? wqb : (blockIdx.z == 1) ? wkb : wvb;
  unsigned short* out = qkv_out + (size_t)blockIdx.z * ((size_t)BATCH * SEQ * D_MODEL);

  v8f acc[2][2];
  for (int i = 0; i < 2; ++i)
    for (int j = 0; j < 2; ++j) acc[i][j] = vzero8();

  const unsigned short* Abase = xb + (size_t)rowBase * D_MODEL;
  const unsigned short* Wbase = W + (size_t)colBase * D_MODEL;

  if (wave == 0) {   // prologue: stage K-step 0 into buffer 0
    tdm_load_2d(lds_offset_of(&bufA[0][0]), Abase, 64, 128, D_MODEL, BATCH * SEQ, D_MODEL);
    tdm_load_2d(lds_offset_of(&bufW[0][0]), Wbase, 64, 64, D_MODEL, D_MODEL, D_MODEL);
  }

  int parity = 0;
  for (int kb = 0; kb < D_MODEL; kb += 64, parity ^= 1) {
    if (wave == 0) {
      if (kb + 64 < D_MODEL) {   // prefetch next K-step into other buffer
        tdm_load_2d(lds_offset_of(&bufA[parity ^ 1][0]), Abase + kb + 64,
                    64, 128, D_MODEL, BATCH * SEQ, D_MODEL);
        tdm_load_2d(lds_offset_of(&bufW[parity ^ 1][0]), Wbase + kb + 64,
                    64, 64, D_MODEL, D_MODEL, D_MODEL);
        __builtin_amdgcn_s_wait_tensorcnt(2);   // current buffer complete
      } else {
        __builtin_amdgcn_s_wait_tensorcnt(0);
      }
    }
    __syncthreads();
    const unsigned short* At = bufA[parity];
    const unsigned short* Wt = bufW[parity];
    for (int ks = 0; ks < 2; ++ks) {
      v16bf aF[2], bF[2];
      for (int mt = 0; mt < 2; ++mt)
        aF[mt] = load_a_frag(At + (wm * 32 + mt * 16) * 64 + ks * 32, 64, lane);
      for (int nt = 0; nt < 2; ++nt)
        bF[nt] = load_b_frag(Wt + (wn * 32 + nt * 16) * 64 + ks * 32, 64, lane);
      for (int mt = 0; mt < 2; ++mt)
        for (int nt = 0; nt < 2; ++nt)
          acc[mt][nt] = __builtin_amdgcn_wmma_f32_16x16x32_bf16(
              false, aF[mt], false, bF[nt], (short)0, acc[mt][nt], false, false);
    }
    __syncthreads();
  }

  const int halfsel = lane >> 4;
  const int nlane   = lane & 15;
  for (int mt = 0; mt < 2; ++mt)
    for (int nt = 0; nt < 2; ++nt)
      for (int r = 0; r < 8; ++r) {
        int row = rowBase + wm * 32 + mt * 16 + r + 8 * halfsel;
        int col = colBase + wn * 32 + nt * 16 + nlane;
        out[(size_t)row * D_MODEL + col] = f2bf(acc[mt][nt][r]);
      }
}

// ---------------------------------------------------------------------------
// Kernel 2: causal flash attention. One wave per 16-row q-tile; key tiles of
// 32. K fragments stream straight from global (two 16B reads per lane, k offsets 8h and 16+8h);
// V is transposed through LDS; P round-trips through LDS to become A frags.
// ---------------------------------------------------------------------------
__global__ __launch_bounds__(128) void attention_kernel(
    const unsigned short* __restrict__ qb,
    const unsigned short* __restrict__ kbuf,
    const unsigned short* __restrict__ vbuf,
    unsigned short* __restrict__ ctx) {
  __shared__ unsigned short vt[4][HDIM * 32];   // V^T per wave: [hd][key]
  __shared__ unsigned short pT[4][16 * 32];     // P per wave:  [qrow][key]
  const int lane = threadIdx.x & 31;
  const int wave = threadIdx.x >> 5;
  const int b  = blockIdx.z;
  const int h  = blockIdx.y;
  const int qt = blockIdx.x * 4 + wave;
  const int q0 = qt * 16;
  unsigned short* myVt = vt[wave];
  unsigned short* myP  = pT[wave];
  const size_t headOff = (size_t)h * HDIM;
  const size_t baseBS  = (size_t)b * SEQ;

  // Preload Q fragments for both 32-wide K-dim steps (HD = 64)
  v16bf qF[2];
  {
    const int m = lane & 15, half = lane >> 4;
    const unsigned short* qrow = qb + (baseBS + q0 + m) * D_MODEL + headOff;
    for (int ks = 0; ks < 2; ++ks) {
      Frag f;
      const unsigned short* p = qrow + ks * 32 + half * 8;
      f.u[0] = *(const u32x4*)(p);
      f.u[1] = *(const u32x4*)(p + 16);
      qF[ks] = f.v;
    }
  }

  v8f accv[4];
  for (int i = 0; i < 4; ++i) accv[i] = vzero8();
  float mrow[8], lrow[8];
  for (int r = 0; r < 8; ++r) { mrow[r] = -1e30f; lrow[r] = 0.0f; }

  const int nlane   = lane & 15;
  const int halfsel = lane >> 4;
  const int ktEnd   = (q0 + 15) / 32 + 1;

  for (int kt = 0; kt < ktEnd; ++kt) {
    const int keyBase = kt * 32;

    // V tile (32 keys x 64 hd) -> LDS transposed
    {
      union { u32x4 q; unsigned short s[8]; } t;
      for (int i = 0; i < 8; ++i) {
        int chunk = i * 32 + lane;           // 256 chunks of 8 elements
        int key   = chunk >> 3;
        int hd0   = (chunk & 7) * 8;
        t.q = *(const u32x4*)(vbuf + (baseBS + keyBase + key) * D_MODEL + headOff + hd0);
        for (int j = 0; j < 8; ++j)
          myVt[(hd0 + j) * 32 + key] = t.s[j];
      }
    }

    // scores = Q K^T  (2 key n-tiles x 2 k-steps)
    v8f s[2];
    s[0] = vzero8();
    s[1] = vzero8();
    for (int ks = 0; ks < 2; ++ks)
      for (int nt = 0; nt < 2; ++nt) {
        Frag f;
        const int kc = halfsel * 8;
        const unsigned short* p =
            kbuf + (baseBS + keyBase + nt * 16 + nlane) * D_MODEL + headOff + ks * 32 + kc;
        f.u[0] = *(const u32x4*)(p);
        f.u[1] = *(const u32x4*)(p + 16);
        s[nt] = __builtin_amdgcn_wmma_f32_16x16x32_bf16(
            false, qF[ks], false, f.v, (short)0, s[nt], false, false);
      }

    // causal mask + online softmax (rows split across 16-lane halves)
    const bool partial = (keyBase + 31 > q0);
    for (int r = 0; r < 8; ++r) {
      const int qrow = q0 + r + 8 * halfsel;
      float s0 = s[0][r] * 0.125f;           // 1/sqrt(64)
      float s1 = s[1][r] * 0.125f;
      if (partial) {
        if (keyBase + nlane > qrow)      s0 = -1e30f;
        if (keyBase + 16 + nlane > qrow) s1 = -1e30f;
      }
      float mx = fmaxf(s0, s1);
      for (int off = 1; off < 16; off <<= 1)
        mx = fmaxf(mx, __shfl_xor(mx, off, 16));
      const float mnew = fmaxf(mrow[r], mx);
      const float sf   = __expf(mrow[r] - mnew);
      const float p0   = __expf(s0 - mnew);
      const float p1   = __expf(s1 - mnew);
      float rs = p0 + p1;
      for (int off = 1; off < 16; off <<= 1)
        rs += __shfl_xor(rs, off, 16);
      lrow[r] = lrow[r] * sf + rs;
      mrow[r] = mnew;
      for (int nt = 0; nt < 4; ++nt) accv[nt][r] *= sf;
      const int prow = r + 8 * halfsel;
      myP[prow * 32 + nlane]      = f2bf(p0);
      myP[prow * 32 + 16 + nlane] = f2bf(p1);
    }

    // ctx += P @ V  (A = P 16x32, B = V^T rows from LDS, 4 hd n-tiles)
    v16bf aP = load_a_frag(myP, 32, lane);
    for (int nt = 0; nt < 4; ++nt) {
      v16bf bV = load_b_frag(myVt + nt * 16 * 32, 32, lane);
      accv[nt] = __builtin_amdgcn_wmma_f32_16x16x32_bf16(
          false, aP, false, bV, (short)0, accv[nt], false, false);
    }
  }

  // normalize and store ctx (bf16, [B,S,D] with head offset)
  for (int r = 0; r < 8; ++r) {
    const float inv = 1.0f / lrow[r];
    const int row = q0 + r + 8 * halfsel;
    for (int nt = 0; nt < 4; ++nt)
      ctx[(baseBS + row) * D_MODEL + headOff + nt * 16 + nlane] =
          f2bf(accv[nt][r] * inv);
  }
}

// ---------------------------------------------------------------------------
// Kernel 3: out = ctx @ Wo^T + bo (bf16 in, fp32 out), TDM-staged tiles.
// ---------------------------------------------------------------------------
__global__ __launch_bounds__(256) void out_proj_kernel(
    const unsigned short* __restrict__ ctx,
    const unsigned short* __restrict__ wob,
    const float* __restrict__ bo, float* __restrict__ out) {
  __shared__ unsigned short bufA[2][128 * 64];
  __shared__ unsigned short bufW[2][64 * 64];
  const int tid  = threadIdx.x;
  const int lane = tid & 31;
  const int wave = tid >> 5;
  const int wm   = wave & 3;
  const int wn   = wave >> 2;
  const int rowBase = blockIdx.x * 128;
  const int colBase = blockIdx.y * 64;

  v8f acc[2][2];
  for (int i = 0; i < 2; ++i)
    for (int j = 0; j < 2; ++j) acc[i][j] = vzero8();

  const unsigned short* Abase = ctx + (size_t)rowBase * D_MODEL;
  const unsigned short* Wbase = wob + (size_t)colBase * D_MODEL;

  if (wave == 0) {
    tdm_load_2d(lds_offset_of(&bufA[0][0]), Abase, 64, 128, D_MODEL, BATCH * SEQ, D_MODEL);
    tdm_load_2d(lds_offset_of(&bufW[0][0]), Wbase, 64, 64, D_MODEL, D_MODEL, D_MODEL);
  }

  int parity = 0;
  for (int kb = 0; kb < D_MODEL; kb += 64, parity ^= 1) {
    if (wave == 0) {
      if (kb + 64 < D_MODEL) {
        tdm_load_2d(lds_offset_of(&bufA[parity ^ 1][0]), Abase + kb + 64,
                    64, 128, D_MODEL, BATCH * SEQ, D_MODEL);
        tdm_load_2d(lds_offset_of(&bufW[parity ^ 1][0]), Wbase + kb + 64,
                    64, 64, D_MODEL, D_MODEL, D_MODEL);
        __builtin_amdgcn_s_wait_tensorcnt(2);
      } else {
        __builtin_amdgcn_s_wait_tensorcnt(0);
      }
    }
    __syncthreads();
    const unsigned short* At = bufA[parity];
    const unsigned short* Wt = bufW[parity];
    for (int ks = 0; ks < 2; ++ks) {
      v16bf aF[2], bF[2];
      for (int mt = 0; mt < 2; ++mt)
        aF[mt] = load_a_frag(At + (wm * 32 + mt * 16) * 64 + ks * 32, 64, lane);
      for (int nt = 0; nt < 2; ++nt)
        bF[nt] = load_b_frag(Wt + (wn * 32 + nt * 16) * 64 + ks * 32, 64, lane);
      for (int mt = 0; mt < 2; ++mt)
        for (int nt = 0; nt < 2; ++nt)
          acc[mt][nt] = __builtin_amdgcn_wmma_f32_16x16x32_bf16(
              false, aF[mt], false, bF[nt], (short)0, acc[mt][nt], false, false);
    }
    __syncthreads();
  }

  const int halfsel = lane >> 4;
  const int nlane   = lane & 15;
  for (int mt = 0; mt < 2; ++mt)
    for (int nt = 0; nt < 2; ++nt)
      for (int r = 0; r < 8; ++r) {
        int row = rowBase + wm * 32 + mt * 16 + r + 8 * halfsel;
        int col = colBase + wn * 32 + nt * 16 + nlane;
        out[(size_t)row * D_MODEL + col] = acc[mt][nt][r] + bo[col];
      }
}

extern "C" void kernel_launch(void* const* d_in, const int* in_sizes, int n_in,
                              void* d_out, int out_size, void* d_ws, size_t ws_size,
                              hipStream_t stream) {
  const float* x  = (const float*)d_in[0];
  const float* Wq = (const float*)d_in[1];
  const float* Wk = (const float*)d_in[2];
  const float* Wv = (const float*)d_in[3];
  const float* Wo = (const float*)d_in[4];
  const float* bo = (const float*)d_in[5];
  float* out = (float*)d_out;

  unsigned short* ws = (unsigned short*)d_ws;
  const size_t N = (size_t)BATCH * SEQ * D_MODEL;   // 4M elements
  const size_t M = (size_t)D_MODEL * D_MODEL;       // 1M elements
  unsigned short* xb   = ws;
  unsigned short* qb   = ws + N;        // qkv_out base (q, k, v back-to-back)
  unsigned short* kbuf = ws + 2 * N;
  unsigned short* vbuf = ws + 3 * N;
  unsigned short* ctxb = ws + 4 * N;
  unsigned short* wqb  = ws + 5 * N;
  unsigned short* wkb  = ws + 5 * N + M;
  unsigned short* wvb  = ws + 5 * N + 2 * M;
  unsigned short* wob  = ws + 5 * N + 3 * M;

  // fp32 -> bf16 staging
  cvt_bf16_kernel<<<1024, 256, 0, stream>>>(x, xb, (int)N);
  cvt_bf16_kernel<<<512, 256, 0, stream>>>(Wq, wqb, (int)M);
  cvt_bf16_kernel<<<512, 256, 0, stream>>>(Wk, wkb, (int)M);
  cvt_bf16_kernel<<<512, 256, 0, stream>>>(Wv, wvb, (int)M);
  cvt_bf16_kernel<<<512, 256, 0, stream>>>(Wo, wob, (int)M);

  dim3 g1((BATCH * SEQ) / 128, D_MODEL / 64, 3);
  qkv_proj_kernel<<<g1, 256, 0, stream>>>(xb, wqb, wkb, wvb, qb);

  dim3 g2(SEQ / 16 / 4, NHEAD, BATCH);
  attention_kernel<<<g2, 128, 0, stream>>>(qb, kbuf, vbuf, ctxb);

  dim3 g3((BATCH * SEQ) / 128, D_MODEL / 64, 1);
  out_proj_kernel<<<g3, 256, 0, stream>>>(ctxb, wob, bo, out);
}
